// Decoder_54949811585417
// MI455X (gfx1250) — matrix operand order repaired and checked by reading, not compiled, not run on hardware
//
#include <hip/hip_runtime.h>
#include <hip/hip_bf16.h>
#include <math.h>

// ---------------------------------------------------------------------------
// Problem constants (from reference): B=1024, T=50, M=128, P=128, F=64
// ---------------------------------------------------------------------------
#define B_ 1024
#define T_ 50
#define M_ 128
#define P_ 128
#define F_ 64
#define G_ 512          // 4*P
#define BN_EPS_ 1e-5f

typedef _Float16 half_t;
typedef __attribute__((ext_vector_type(16))) _Float16 v16h;
typedef __attribute__((ext_vector_type(8)))  float    v8f;
typedef __attribute__((ext_vector_type(4)))  float    f32x4;

union AFrag { v16h v; f32x4 f4[2]; };
union H8    { f32x4 f; half_t h[8]; };

// A-matrix fragment (16x32 f16), row-major source with leading dim ld (halfs).
// ISA layout: lanes 0-15 -> M=lane, halfs = K {k0..k0+7, k0+16..k0+23};
//             lanes 16-31 -> M=lane-16, halfs = K {k0+8.., k0+24..}.
__device__ __forceinline__ v16h load_frag_a(const half_t* base, int ld, int k0) {
    int lane = threadIdx.x & 31;
    int row  = lane & 15;
    int ko   = (lane >> 4) << 3;           // 0 or 8
    const half_t* p = base + row * ld + k0 + ko;
    AFrag u;
    u.f4[0] = *(const f32x4*)(p);          // K = k0+ko .. +7
    u.f4[1] = *(const f32x4*)(p + 16);     // K = k0+ko+16 .. +7
    return u.v;
}

// B-matrix fragment (32x16 f16), K-major source W[k][n] with leading dim ldn.
// ISA layout: lane L holds column N = n0+L%16, halfs = K {k0+ko..+7, k0+ko+16..+23}, ko = 8*(L/16).
__device__ __forceinline__ v16h load_frag_b(const half_t* W, int ldn, int k0, int n0) {
    int lane = threadIdx.x & 31;
    int ko   = (lane >> 4) << 3;
    const half_t* p = W + (size_t)(k0 + ko) * ldn + n0 + (lane & 15);
    v16h v;
#pragma unroll
    for (int s = 0; s < 8; ++s) {
        v[s]     = p[(size_t)s * ldn];
        v[s + 8] = p[(size_t)(s + 16) * ldn];
    }
    return v;
}

__device__ __forceinline__ v8f wmma_f16(v16h a, v16h b, v8f c) {
    return __builtin_amdgcn_wmma_f32_16x16x32_f16(false, a, false, b, (short)0, c, false, false);
}

__device__ __forceinline__ float sigmoidf_(float x) { return 1.0f / (1.0f + expf(-x)); }

// ---------------------------------------------------------------------------
// Kernel 1: f32 -> f16 conversion of X, and K-major f16 weight transposes.
// ---------------------------------------------------------------------------
__global__ void prep_kernel(const float* __restrict__ X,
                            const float* __restrict__ attn_w1,
                            const float* __restrict__ fc_w,
                            const float* __restrict__ w_ih0, const float* __restrict__ w_hh0,
                            const float* __restrict__ w_ih1, const float* __restrict__ w_hh1,
                            const float* __restrict__ fcf_w,
                            half_t* __restrict__ Xh,
                            half_t* __restrict__ Wd, half_t* __restrict__ Wc, half_t* __restrict__ Wx,
                            half_t* __restrict__ Fc,
                            half_t* __restrict__ Wih0, half_t* __restrict__ Whh0,
                            half_t* __restrict__ Wih1, half_t* __restrict__ Whh1,
                            half_t* __restrict__ Fcf) {
    size_t i0 = (size_t)blockIdx.x * blockDim.x + threadIdx.x;
    size_t stride = (size_t)gridDim.x * blockDim.x;
    for (size_t i = i0; i < (size_t)B_ * T_ * M_; i += stride) Xh[i] = (half_t)X[i];
    for (size_t i = i0; i < (size_t)M_ * P_; i += stride) {
        int p = (int)(i >> 7), n = (int)(i & 127);
        // attn_w1: (M=128 rows n) x (384 cols = [d:128 | c:128 | x:128])
        Wd[p * M_ + n] = (half_t)attn_w1[n * 384 + p];
        Wc[p * M_ + n] = (half_t)attn_w1[n * 384 + 128 + p];
        Wx[p * M_ + n] = (half_t)attn_w1[n * 384 + 256 + p];
    }
    for (size_t i = i0; i < (size_t)192 * F_; i += stride) {
        int k = (int)(i >> 6), n = (int)(i & 63);
        Fc[k * F_ + n] = (half_t)fc_w[n * 192 + k];
    }
    for (size_t i = i0; i < (size_t)F_ * G_; i += stride) {
        int k = (int)(i >> 9), n = (int)(i & 511);
        Wih0[k * G_ + n] = (half_t)w_ih0[n * F_ + k];
    }
    for (size_t i = i0; i < (size_t)P_ * G_; i += stride) {
        int k = (int)(i >> 9), n = (int)(i & 511);
        Whh0[k * G_ + n] = (half_t)w_hh0[n * P_ + k];
        Wih1[k * G_ + n] = (half_t)w_ih1[n * P_ + k];
        Whh1[k * G_ + n] = (half_t)w_hh1[n * P_ + k];
    }
    for (size_t i = i0; i < (size_t)256 * F_; i += stride) {
        int k = (int)(i >> 6), n = (int)(i & 63);
        Fcf[k * F_ + n] = (half_t)fcf_w[n * 256 + k];
    }
}

// ---------------------------------------------------------------------------
// Kernel 2: x_proj[bt, n] = sum_m Xh[bt, m] * Wx[m, n]   (51200 x 128 x 128)
// 16-row tile per block, 8 wave32s -> one 16-col N tile per wave.
// ---------------------------------------------------------------------------
__global__ void xproj_kernel(const half_t* __restrict__ Xh,
                             const half_t* __restrict__ Wx,
                             half_t* __restrict__ XPh) {
    int bt0  = blockIdx.x * 16;
    int wv   = threadIdx.x >> 5;
    int lane = threadIdx.x & 31;
    int n0   = wv * 16;
    v8f acc = {};
    const half_t* Abase = Xh + (size_t)bt0 * M_;
#pragma unroll
    for (int k0 = 0; k0 < M_; k0 += 32) {
        v16h a = load_frag_a(Abase, M_, k0);
        v16h b = load_frag_b(Wx, M_, k0, n0);
        acc = wmma_f16(a, b, acc);
    }
    int crow = (lane >> 4) << 3;
    int ccol = lane & 15;
#pragma unroll
    for (int r = 0; r < 8; ++r)
        XPh[(size_t)(bt0 + crow + r) * M_ + n0 + ccol] = (half_t)acc[r];
}

// ---------------------------------------------------------------------------
// Kernel 3: full recurrent decoder. One block = 16 batch rows, 256 threads
// (8 wave32s). State lives in LDS across all T=50 steps.
// ---------------------------------------------------------------------------
#define SMEM_BYTES 82432

__global__ void __launch_bounds__(256)
decoder_kernel(const float* __restrict__ yprev,
               const half_t* __restrict__ Xh,  const half_t* __restrict__ XPh,
               const half_t* __restrict__ Wd,  const half_t* __restrict__ Wc,
               const half_t* __restrict__ Fc,
               const half_t* __restrict__ Wih0, const half_t* __restrict__ Whh0,
               const half_t* __restrict__ Wih1, const half_t* __restrict__ Whh1,
               const half_t* __restrict__ Fcf,
               const float* __restrict__ attn_b1, const float* __restrict__ attn_w2,
               const float* __restrict__ attn_b2,
               const float* __restrict__ fc_b,
               const float* __restrict__ bn_gamma, const float* __restrict__ bn_beta,
               const float* __restrict__ bn_mean,  const float* __restrict__ bn_var,
               const float* __restrict__ b_ih0, const float* __restrict__ b_hh0,
               const float* __restrict__ b_ih1, const float* __restrict__ b_hh1,
               const float* __restrict__ fcf_b,
               float* __restrict__ out) {
    extern __shared__ char smem[];
    half_t* h1h  = (half_t*)(smem);             // 16x128 f16
    half_t* c1h  = (half_t*)(smem + 4096);      // 16x128 f16
    half_t* h0h  = (half_t*)(smem + 8192);      // 16x128 f16
    half_t* ctxh = (half_t*)(smem + 12288);     // 16x128 f16
    half_t* yth  = (half_t*)(smem + 16384);     // 16x64  f16
    half_t* ytl  = (half_t*)(smem + 18432);     // 16x64  f16 (y_tilde)
    float*  sp   = (float*) (smem + 20480);     // 16x128 f32 state_proj
    float*  c0   = (float*) (smem + 28672);     // 16x128 f32
    float*  c1   = (float*) (smem + 36864);     // 16x128 f32
    float*  gates= (float*) (smem + 45056);     // 16x512 f32
    float*  sc   = (float*) (smem + 77824);     // 16x64  f32 score/beta
    float*  w2s  = (float*) (smem + 81920);     // 128    f32

    const int tid  = threadIdx.x;
    const int wv   = tid >> 5;
    const int lane = tid & 31;
    const int crow = (lane >> 4) << 3;
    const int ccol = lane & 15;
    const int b0   = blockIdx.x * 16;

    // zero initial state
    for (int i = tid; i < 16 * P_; i += 256) {
        h1h[i] = (half_t)0.f; c1h[i] = (half_t)0.f; h0h[i] = (half_t)0.f;
        c0[i] = 0.f; c1[i] = 0.f;
    }
    if (tid < M_) w2s[tid] = attn_w2[tid];
    __syncthreads();

    for (int t = 0; t < T_; ++t) {
        // warm next step's y_prev slice into cache while this step computes
        if (t + 1 < T_) {
            int i = tid;                           // 256 threads x2 cover rows 0-7 (8*64 floats)
            int bl = i >> 6, f = i & 63;
            __builtin_prefetch(&yprev[((size_t)(b0 + bl) * T_ + (t + 1)) * F_ + f], 0, 3);
            bl = (i + 256) >> 6; f = (i + 256) & 63;
            __builtin_prefetch(&yprev[((size_t)(b0 + bl) * T_ + (t + 1)) * F_ + f], 0, 3);
        }

        // ---- phase 1: state_proj = h1 @ Wd + c1 @ Wc + b1   (16x128) -------
        {
            int n0 = wv * 16;
            v8f acc = {};
#pragma unroll
            for (int k0 = 0; k0 < P_; k0 += 32)
                acc = wmma_f16(load_frag_a(h1h, P_, k0), load_frag_b(Wd, M_, k0, n0), acc);
#pragma unroll
            for (int k0 = 0; k0 < P_; k0 += 32)
                acc = wmma_f16(load_frag_a(c1h, P_, k0), load_frag_b(Wc, M_, k0, n0), acc);
            float bn = attn_b1[n0 + ccol];
#pragma unroll
            for (int r = 0; r < 8; ++r)
                sp[(crow + r) * M_ + n0 + ccol] = acc[r] + bn;
        }
        // stage y_t into LDS as f16
        for (int i = tid; i < 16 * F_; i += 256) {
            int bl = i / F_, f = i % F_;
            yth[i] = (half_t)yprev[((size_t)(b0 + bl) * T_ + t) * F_ + f];
        }
        __syncthreads();

        // ---- phase 2: score[b,t'] = sum_n tanh(xp + sp) * w2 + b2 ----------
        {
            int bl = tid >> 4;
            const float* sprow = sp + bl * M_;
            for (int tt = (tid & 15); tt < T_; tt += 16) {
                const half_t* xp = XPh + ((size_t)(b0 + bl) * T_ + tt) * M_;
                if (tt + 16 < T_)   // prefetch next row for this thread
                    __builtin_prefetch(xp + 16 * M_, 0, 3);
                float s = 0.f;
                for (int n = 0; n < M_; n += 8) {
                    H8 u; u.f = *(const f32x4*)(xp + n);
#pragma unroll
                    for (int k = 0; k < 8; ++k)
                        s += tanhf((float)u.h[k] + sprow[n + k]) * w2s[n + k];
                }
                sc[bl * 64 + tt] = s + attn_b2[0];
            }
        }
        __syncthreads();

        // ---- phase 3: softmax over T per batch row --------------------------
        if (tid < 16) {
            float mx = -1e30f;
            for (int tt = 0; tt < T_; ++tt) mx = fmaxf(mx, sc[tid * 64 + tt]);
            float sm = 0.f;
            for (int tt = 0; tt < T_; ++tt) {
                float e = expf(sc[tid * 64 + tt] - mx);
                sc[tid * 64 + tt] = e; sm += e;
            }
            float inv = 1.f / sm;
            for (int tt = 0; tt < T_; ++tt) sc[tid * 64 + tt] *= inv;
        }
        __syncthreads();

        // ---- phase 4: context[b,m] = sum_t beta * X[b,t,m] ------------------
        {
            int bl = tid >> 4, m0 = (tid & 15) * 8;
            const half_t* xb = Xh + (size_t)(b0 + bl) * T_ * M_ + m0;
            float acc[8] = {0.f,0.f,0.f,0.f,0.f,0.f,0.f,0.f};
            for (int tt = 0; tt < T_; ++tt) {
                float w = sc[bl * 64 + tt];
                if (tt + 1 < T_)
                    __builtin_prefetch(xb + (size_t)(tt + 1) * M_, 0, 3);
                H8 u; u.f = *(const f32x4*)(xb + (size_t)tt * M_);
#pragma unroll
                for (int k = 0; k < 8; ++k) acc[k] += w * (float)u.h[k];
            }
#pragma unroll
            for (int k = 0; k < 8; ++k) ctxh[bl * M_ + m0 + k] = (half_t)acc[k];
        }
        __syncthreads();

        // ---- phase 5: z = [ctx|y_t] @ FcT, then batchnorm -> y_tilde --------
        if (wv < 4) {
            int n0 = wv * 16;
            v8f acc = {};
#pragma unroll
            for (int k0 = 0; k0 < M_; k0 += 32)
                acc = wmma_f16(load_frag_a(ctxh, M_, k0), load_frag_b(Fc, F_, k0, n0), acc);
#pragma unroll
            for (int k0 = 0; k0 < F_; k0 += 32)
                acc = wmma_f16(load_frag_a(yth, F_, k0), load_frag_b(Fc, F_, M_ + k0, n0), acc);
            int n = n0 + ccol;
            float zb = fc_b[n], mean = bn_mean[n];
            float inv = rsqrtf(bn_var[n] + BN_EPS_);
            float gm = bn_gamma[n], be = bn_beta[n];
#pragma unroll
            for (int r = 0; r < 8; ++r) {
                float z = acc[r] + zb;
                ytl[(crow + r) * F_ + n] = (half_t)((z - mean) * inv * gm + be);
            }
        }
        __syncthreads();

        // ---- phase 6: gates0 = y_tilde @ Wih0T + h0 @ Whh0T + b -------------
#pragma unroll
        for (int j = 0; j < 4; ++j) {
            int n0 = 16 * (wv + 8 * j);
            v8f acc = {};
#pragma unroll
            for (int k0 = 0; k0 < F_; k0 += 32)
                acc = wmma_f16(load_frag_a(ytl, F_, k0), load_frag_b(Wih0, G_, k0, n0), acc);
#pragma unroll
            for (int k0 = 0; k0 < P_; k0 += 32)
                acc = wmma_f16(load_frag_a(h0h, P_, k0), load_frag_b(Whh0, G_, k0, n0), acc);
            int n = n0 + ccol;
            float bsum = b_ih0[n] + b_hh0[n];
#pragma unroll
            for (int r = 0; r < 8; ++r)
                gates[(crow + r) * G_ + n] = acc[r] + bsum;
        }
        __syncthreads();

        // ---- LSTM0 elementwise ---------------------------------------------
        for (int i = tid; i < 16 * P_; i += 256) {
            int bl = i >> 7, p = i & 127;
            const float* g = gates + bl * G_;
            float ig = sigmoidf_(g[p]),        fg = sigmoidf_(g[128 + p]);
            float gg = tanhf(g[256 + p]),      og = sigmoidf_(g[384 + p]);
            float cn = fg * c0[i] + ig * gg;
            c0[i] = cn;
            h0h[i] = (half_t)(og * tanhf(cn));
        }
        __syncthreads();

        // ---- phase 7: gates1 = h0 @ Wih1T + h1 @ Whh1T + b ------------------
#pragma unroll
        for (int j = 0; j < 4; ++j) {
            int n0 = 16 * (wv + 8 * j);
            v8f acc = {};
#pragma unroll
            for (int k0 = 0; k0 < P_; k0 += 32)
                acc = wmma_f16(load_frag_a(h0h, P_, k0), load_frag_b(Wih1, G_, k0, n0), acc);
#pragma unroll
            for (int k0 = 0; k0 < P_; k0 += 32)
                acc = wmma_f16(load_frag_a(h1h, P_, k0), load_frag_b(Whh1, G_, k0, n0), acc);
            int n = n0 + ccol;
            float bsum = b_ih1[n] + b_hh1[n];
#pragma unroll
            for (int r = 0; r < 8; ++r)
                gates[(crow + r) * G_ + n] = acc[r] + bsum;
        }
        __syncthreads();

        // ---- LSTM1 elementwise ---------------------------------------------
        for (int i = tid; i < 16 * P_; i += 256) {
            int bl = i >> 7, p = i & 127;
            const float* g = gates + bl * G_;
            float ig = sigmoidf_(g[p]),        fg = sigmoidf_(g[128 + p]);
            float gg = tanhf(g[256 + p]),      og = sigmoidf_(g[384 + p]);
            float cn = fg * c1[i] + ig * gg;
            c1[i] = cn;
            c1h[i] = (half_t)cn;
            h1h[i] = (half_t)(og * tanhf(cn));
        }
        __syncthreads();
    }

    // ---- final: y_pred = relu([h1|ctx] @ FcfT + b) -------------------------
    if (wv < 4) {
        int n0 = wv * 16;
        v8f acc = {};
#pragma unroll
        for (int k0 = 0; k0 < P_; k0 += 32)
            acc = wmma_f16(load_frag_a(h1h, P_, k0), load_frag_b(Fcf, F_, k0, n0), acc);
#pragma unroll
        for (int k0 = 0; k0 < M_; k0 += 32)
            acc = wmma_f16(load_frag_a(ctxh, M_, k0), load_frag_b(Fcf, F_, P_ + k0, n0), acc);
        int n = n0 + ccol;
        float bb = fcf_b[n];
#pragma unroll
        for (int r = 0; r < 8; ++r)
            out[(size_t)(b0 + crow + r) * F_ + n] = fmaxf(acc[r] + bb, 0.f);
    }
}

// ---------------------------------------------------------------------------
// Host launcher
// ---------------------------------------------------------------------------
extern "C" void kernel_launch(void* const* d_in, const int* in_sizes, int n_in,
                              void* d_out, int out_size, void* d_ws, size_t ws_size,
                              hipStream_t stream) {
    const float* X        = (const float*)d_in[0];
    const float* yprev    = (const float*)d_in[1];
    const float* attn_w1  = (const float*)d_in[2];
    const float* attn_b1  = (const float*)d_in[3];
    const float* attn_w2  = (const float*)d_in[4];
    const float* attn_b2  = (const float*)d_in[5];
    const float* fc_w     = (const float*)d_in[6];
    const float* fc_b     = (const float*)d_in[7];
    const float* bn_gamma = (const float*)d_in[8];
    const float* bn_beta  = (const float*)d_in[9];
    const float* bn_mean  = (const float*)d_in[10];
    const float* bn_var   = (const float*)d_in[11];
    const float* w_ih0    = (const float*)d_in[12];
    const float* w_hh0    = (const float*)d_in[13];
    const float* b_ih0    = (const float*)d_in[14];
    const float* b_hh0    = (const float*)d_in[15];
    const float* w_ih1    = (const float*)d_in[16];
    const float* w_hh1    = (const float*)d_in[17];
    const float* b_ih1    = (const float*)d_in[18];
    const float* b_hh1    = (const float*)d_in[19];
    const float* fcf_w    = (const float*)d_in[20];
    const float* fcf_b    = (const float*)d_in[21];
    float* out = (float*)d_out;

    // workspace carve (f16 buffers), 256B aligned
    char* ws = (char*)d_ws;
    size_t off = 0;
    auto carve = [&](size_t bytes) {
        char* p = ws + off;
        off += (bytes + 255) & ~(size_t)255;
        return p;
    };
    half_t* Xh   = (half_t*)carve((size_t)B_ * T_ * M_ * 2);
    half_t* XPh  = (half_t*)carve((size_t)B_ * T_ * M_ * 2);
    half_t* Wd   = (half_t*)carve((size_t)P_ * M_ * 2);
    half_t* Wc   = (half_t*)carve((size_t)P_ * M_ * 2);
    half_t* Wx   = (half_t*)carve((size_t)M_ * M_ * 2);
    half_t* Fc   = (half_t*)carve((size_t)192 * F_ * 2);
    half_t* Wih0 = (half_t*)carve((size_t)F_ * G_ * 2);
    half_t* Whh0 = (half_t*)carve((size_t)P_ * G_ * 2);
    half_t* Wih1 = (half_t*)carve((size_t)P_ * G_ * 2);
    half_t* Whh1 = (half_t*)carve((size_t)P_ * G_ * 2);
    half_t* Fcf  = (half_t*)carve((size_t)256 * F_ * 2);

    prep_kernel<<<dim3(2048), dim3(256), 0, stream>>>(
        X, attn_w1, fc_w, w_ih0, w_hh0, w_ih1, w_hh1, fcf_w,
        Xh, Wd, Wc, Wx, Fc, Wih0, Whh0, Wih1, Whh1, Fcf);

    xproj_kernel<<<dim3(B_ * T_ / 16), dim3(256), 0, stream>>>(Xh, Wx, XPh);

    decoder_kernel<<<dim3(B_ / 16), dim3(256), SMEM_BYTES, stream>>>(
        yprev, Xh, XPh, Wd, Wc, Fc, Wih0, Whh0, Wih1, Whh1, Fcf,
        attn_b1, attn_w2, attn_b2, fc_b,
        bn_gamma, bn_beta, bn_mean, bn_var,
        b_ih0, b_hh0, b_ih1, b_hh1, fcf_b, out);
}
